// GraphAttentionLayer2_32023276159121
// MI455X (gfx1250) — matrix operand order repaired and checked by reading, not compiled, not run on hardware
//
#include <hip/hip_runtime.h>
#include <hip/hip_bf16.h>

// ---- types ----
typedef __attribute__((ext_vector_type(16))) _Float16 v16h;
typedef __attribute__((ext_vector_type(8)))  _Float16 v8h;
typedef __attribute__((ext_vector_type(8)))  float    v8f;
typedef __attribute__((ext_vector_type(8)))  float    f8v;

#define ALPHA   0.2f
#define NEG_INF (-9.0e15f)

// problem constants
#define BT   48
#define NN   1024
#define CIN  64
#define FOUT 64
#define EM   16

// LDS row stride for staged WhT (pad 8 halves -> 2064 B -> 4-dword bank step)
#define LDS_STRIDE 1032

// Async global->LDS staging if the gfx1250 builtins are available.
#if defined(__has_builtin)
#if __has_builtin(__builtin_amdgcn_global_load_async_to_lds_b128) && \
    __has_builtin(__builtin_amdgcn_s_wait_asynccnt)
#define USE_ASYNC_LDS 1
#endif
#endif
#ifndef USE_ASYNC_LDS
#define USE_ASYNC_LDS 0
#endif

#if USE_ASYNC_LDS
// Builtin signature (from hipcc diagnostic): param0 is non-const
// 'int __attribute__((vector_size(16))) __device__ *'; 4 args total.
typedef int v4i_vs __attribute__((vector_size(16)));
typedef __attribute__((address_space(1))) v4i_vs GlobV4;   // global (__device__)
typedef __attribute__((address_space(3))) v4i_vs LdsV4;    // LDS (__shared__)
#endif

// workspace layout (bytes)
//   WhT    : _Float16 [BT][FOUT][NN]   = 6,291,456 B
//   rterm  : float    [BT*NN]          =   196,608 B
//   cterm  : float    [BT*NN]          =   196,608 B
//   adjbits: uint     [NN][NN/32]      =   131,072 B
#define WS_WHT_OFF   ((size_t)0)
#define WS_RTERM_OFF ((size_t)(BT * FOUT * NN) * sizeof(_Float16))
#define WS_CTERM_OFF (WS_RTERM_OFF + (size_t)(BT * NN) * sizeof(float))
#define WS_ADJB_OFF  (WS_CTERM_OFF + (size_t)(BT * NN) * sizeof(float))

static __device__ __forceinline__ float lane_bcast(float v, int srcLane) {
    int i = __builtin_bit_cast(int, v);
    int r = __builtin_amdgcn_ds_bpermute(srcLane << 2, i);   // wave32: index bits [6:2]
    return __builtin_bit_cast(float, r);
}

// =====================================================================
// Kernel 0: pack adjacency into bitmask. adjbits[i][c] bit e = adj[i][32c+e] > 0
// =====================================================================
__global__ __launch_bounds__(256) void gat_adjpack_kernel(
    const int* __restrict__ adj, unsigned* __restrict__ adjbits)
{
    const int idx = blockIdx.x * 256 + threadIdx.x;   // [0, NN*32)
    const int i = idx >> 5, c = idx & 31;
    const int* p = adj + (size_t)i * NN + c * 32;
    unsigned w = 0;
#pragma unroll
    for (int e = 0; e < 32; ++e) w |= (p[e] > 0 ? 1u : 0u) << e;
    adjbits[idx] = w;
}

// =====================================================================
// Kernel 1: Wh = x @ W, stored transposed as f16: WhT[bt][f][n]
// W's B-fragments are built once per block into LDS (shared by 8 waves).
// =====================================================================
__global__ __launch_bounds__(256) void gat_proj_kernel(
    const float* __restrict__ x,      // [BT*NN, CIN]
    const float* __restrict__ W,      // [CIN, FOUT]
    _Float16*    __restrict__ whT)    // [BT][FOUT][NN]
{
    // LDS fragment buffer: [frag q=0..7][half 0..1][lane 0..31][8 halves] = 8 KB
    __shared__ _Float16 wfr[8 * 2 * 32 * 8];

    const int tid  = threadIdx.x;
    const int lane = tid & 31;
    const int wv   = tid >> 5;
    const int li   = lane & 15;
    const bool hi  = lane >= 16;

    // ---- cooperative build of the 8 B-fragments of W (f32 -> f16) ----
    {
        const int fs   = tid >> 7;          // k-step of this fill slot
        const int ft   = (tid >> 5) & 3;    // col tile
        const int q    = fs * 4 + ft;
        const int col  = ft * 16 + li;
        const int kb   = fs * 32 + (hi ? 8 : 0);
        v8h h0, h1;
#pragma unroll
        for (int e = 0; e < 8; ++e) {
            h0[e] = (_Float16)W[(size_t)(kb + e) * FOUT + col];
            h1[e] = (_Float16)W[(size_t)(kb + 16 + e) * FOUT + col];
        }
        *(v8h*)(wfr + ((q * 2 + 0) * 32 + lane) * 8) = h0;
        *(v8h*)(wfr + ((q * 2 + 1) * 32 + lane) * 8) = h1;
    }
    __syncthreads();

    const int rowBase = blockIdx.x * 128 + wv * 16;   // global row in [0, BT*NN)
    const int bt      = rowBase >> 10;
    const int nBase   = rowBase & (NN - 1);

    const float* xrow = x + (size_t)(rowBase + li) * CIN;
    const int akb = hi ? 8 : 0;                        // A-fragment K base

    v8f acc0 = {}, acc1 = {}, acc2 = {}, acc3 = {};

#pragma unroll
    for (int s = 0; s < 2; ++s) {                      // K steps of 32
        // ---- A fragment: rows of x, f32 -> f16 ----
        const int k0 = s * 32 + akb;
        f8v x0 = *(const f8v*)(xrow + k0);             // K = k0 .. k0+7
        f8v x1 = *(const f8v*)(xrow + k0 + 16);        // K = k0+16 .. k0+23
        v16h a;
#pragma unroll
        for (int e = 0; e < 8; ++e) {
            a[e]     = (_Float16)x0[e];
            a[8 + e] = (_Float16)x1[e];
        }
        // ---- B fragments from LDS (conflict-free 16B reads) ----
        v16h b[4];
#pragma unroll
        for (int t = 0; t < 4; ++t) {
            const int q = s * 4 + t;
            v8h h0 = *(const v8h*)(wfr + ((q * 2 + 0) * 32 + lane) * 8);
            v8h h1 = *(const v8h*)(wfr + ((q * 2 + 1) * 32 + lane) * 8);
#pragma unroll
            for (int e = 0; e < 8; ++e) { b[t][e] = h0[e]; b[t][8 + e] = h1[e]; }
        }
        acc0 = __builtin_amdgcn_wmma_f32_16x16x32_f16(false, a, false, b[0], (short)0, acc0, false, false);
        acc1 = __builtin_amdgcn_wmma_f32_16x16x32_f16(false, a, false, b[1], (short)0, acc1, false, false);
        acc2 = __builtin_amdgcn_wmma_f32_16x16x32_f16(false, a, false, b[2], (short)0, acc2, false, false);
        acc3 = __builtin_amdgcn_wmma_f32_16x16x32_f16(false, a, false, b[3], (short)0, acc3, false, false);
    }

    // ---- transposed f16 store: WhT[(bt*64 + f)*1024 + n] ----
    const int rowoff = hi ? 8 : 0;
#pragma unroll
    for (int v = 0; v < 8; ++v) {
        const int n = nBase + v + rowoff;
        _Float16* dst = whT + ((size_t)bt * FOUT + li) * NN + n;
        dst[(size_t)0  * NN] = (_Float16)acc0[v];
        dst[(size_t)16 * NN] = (_Float16)acc1[v];
        dst[(size_t)32 * NN] = (_Float16)acc2[v];
        dst[(size_t)48 * NN] = (_Float16)acc3[v];
    }
}

// =====================================================================
// Kernel 2: per-(bt,n) score terms
//   rterm = Wh[bt,n,:]. a[:64]  + emb1[n]. a2[:16]
//   cterm = Wh[bt,n,:]. a[64:]  + emb2[n]. a2[16:]
// =====================================================================
__global__ __launch_bounds__(256) void gat_scores_kernel(
    const _Float16* __restrict__ whT,
    const float* __restrict__ emb1, const float* __restrict__ emb2,
    const float* __restrict__ a,    const float* __restrict__ a2,
    float* __restrict__ rterm, float* __restrict__ cterm)
{
    const int idx = blockIdx.x * 256 + threadIdx.x;    // [0, BT*NN)
    const int bt = idx >> 10;
    const int n  = idx & (NN - 1);

    const _Float16* wh = whT + (size_t)bt * FOUT * NN + n;
    float s1 = 0.0f, s2 = 0.0f;
#pragma unroll
    for (int f = 0; f < FOUT; ++f) {
        float w = (float)wh[(size_t)f * NN];
        s1 = fmaf(w, a[f], s1);
        s2 = fmaf(w, a[FOUT + f], s2);
    }
    float f1 = 0.0f, f2 = 0.0f;
#pragma unroll
    for (int e = 0; e < EM; ++e) {
        f1 = fmaf(emb1[n * EM + e], a2[e], f1);
        f2 = fmaf(emb2[n * EM + e], a2[EM + e], f2);
    }
    rterm[idx] = s1 + f1;
    cterm[idx] = s2 + f2;
}

// =====================================================================
// Kernel 3: flash-style masked softmax + (att @ Wh) + ELU
// Block stages WhT[bt] (128 KB) into padded LDS once (async direct-to-LDS
// when available); 8 waves each own a 16-row x 64-col output block.
// =====================================================================
__global__ __launch_bounds__(256) void gat_attn_kernel(
    const unsigned* __restrict__ adjbits, // [NN][32]
    const _Float16* __restrict__ whT,     // [BT][FOUT][NN]
    const float* __restrict__ rterm,
    const float* __restrict__ cterm,
    float* __restrict__ out)              // [BT*NN, FOUT]
{
    extern __shared__ _Float16 whs[];     // [64][LDS_STRIDE] = 129 KB

    const int tid  = threadIdx.x;
    const int lane = tid & 31;
    const int wv   = tid >> 5;
    const int bt   = blockIdx.y;
    const int i0   = blockIdx.x * 128 + wv * 16;
    const int li   = lane & 15;
    const bool hi  = lane >= 16;
    const int kbase = hi ? 8 : 0;         // A-fragment K base (bit offset too)
    const int bkb   = hi ? 8 : 0;         // B-fragment K base

    const _Float16* wh = whT + (size_t)bt * FOUT * NN;

    // ---- stage WhT[bt] into LDS (padded rows), 16B pieces ----
#if USE_ASYNC_LDS
    for (int p = tid; p < FOUT * (NN / 8); p += 256) {   // 8192 pieces of 16B
        const int f   = p >> 7;                          // 128 pieces / row
        const int off = (p & 127) * 8;
        __builtin_amdgcn_global_load_async_to_lds_b128(
            (GlobV4*)(wh + (size_t)f * NN + off),
            (LdsV4*)(whs + (size_t)f * LDS_STRIDE + off),
            0, 0);
    }
    __builtin_amdgcn_s_wait_asynccnt(0);   // drain this wave's async copies
    __syncthreads();                       // publish LDS tile to all waves
#else
    for (int p = tid; p < FOUT * (NN / 8); p += 256) {   // 8192 pieces
        const int f   = p >> 7;                          // 128 pieces / row
        const int off = (p & 127) * 8;
        *(v8h*)(whs + (size_t)f * LDS_STRIDE + off) =
            *(const v8h*)(wh + (size_t)f * NN + off);
    }
    __syncthreads();
#endif

    const float r = rterm[bt * NN + i0 + li];
    const unsigned* abRow = adjbits + (size_t)(i0 + li) * 32;
    const float* cRow = cterm + bt * NN;

    v8f acc0 = {}, acc1 = {}, acc2 = {}, acc3 = {};
    float m = -3.0e38f, l = 0.0f;

    for (int j0 = 0; j0 < NN; j0 += 32) {
        const int jo0 = j0 + kbase;       // 8 contiguous j's
        const int jo1 = jo0 + 16;         // 8 more
        const unsigned aw = abRow[j0 >> 5];
        f8v cc0 = *(const f8v*)(cRow + jo0);
        f8v cc1 = *(const f8v*)(cRow + jo1);

        // ---- masked leaky-relu scores (16 per lane) ----
        float sc[16];
#pragma unroll
        for (int e = 0; e < 8; ++e) {
            float s0 = r + cc0[e];  s0 = (s0 > 0.0f) ? s0 : ALPHA * s0;
            sc[e]     = ((aw >> (kbase + e)) & 1u) ? s0 : NEG_INF;
            float s1 = r + cc1[e];  s1 = (s1 > 0.0f) ? s1 : ALPHA * s1;
            sc[8 + e] = ((aw >> (kbase + 16 + e)) & 1u) ? s1 : NEG_INF;
        }

        // ---- online softmax: row max over 32 j's (lanes li and li+16) ----
        float lm = sc[0];
#pragma unroll
        for (int e = 1; e < 16; ++e) lm = fmaxf(lm, sc[e]);
        lm = fmaxf(lm, lane_bcast(lm, lane ^ 16));
        const float mN    = fmaxf(m, lm);
        const float scale = __expf(m - mN);

        v16h afrag;
        float ls = 0.0f;
#pragma unroll
        for (int e = 0; e < 16; ++e) {
            float p = __expf(sc[e] - mN);
            ls += p;
            afrag[e] = (_Float16)p;
        }
        ls += lane_bcast(ls, lane ^ 16);
        l = l * scale + ls;
        m = mN;

        // ---- rescale accumulators (per-row factors broadcast to col lanes) ----
        const int rowoff = hi ? 8 : 0;
        float fsc[8];
#pragma unroll
        for (int v = 0; v < 8; ++v) fsc[v] = lane_bcast(scale, v + rowoff);
#pragma unroll
        for (int v = 0; v < 8; ++v) {
            acc0[v] *= fsc[v]; acc1[v] *= fsc[v];
            acc2[v] *= fsc[v]; acc3[v] *= fsc[v];
        }

        // ---- B fragments from LDS: K = bkb..bkb+7 and bkb+16..bkb+23 per lane ----
        v16h b[4];
#pragma unroll
        for (int t = 0; t < 4; ++t) {
            const _Float16* bp = whs + (size_t)(li + 16 * t) * LDS_STRIDE + j0 + bkb;
            v8h h0 = *(const v8h*)(bp);
            v8h h1 = *(const v8h*)(bp + 16);
#pragma unroll
            for (int e = 0; e < 8; ++e) { b[t][e] = h0[e]; b[t][8 + e] = h1[e]; }
        }

        acc0 = __builtin_amdgcn_wmma_f32_16x16x32_f16(false, afrag, false, b[0], (short)0, acc0, false, false);
        acc1 = __builtin_amdgcn_wmma_f32_16x16x32_f16(false, afrag, false, b[1], (short)0, acc1, false, false);
        acc2 = __builtin_amdgcn_wmma_f32_16x16x32_f16(false, afrag, false, b[2], (short)0, acc2, false, false);
        acc3 = __builtin_amdgcn_wmma_f32_16x16x32_f16(false, afrag, false, b[3], (short)0, acc3, false, false);
    }

    // ---- finalize: divide by row sums, ELU, store ----
    const int rowoff = hi ? 8 : 0;
    float linv[8];
#pragma unroll
    for (int v = 0; v < 8; ++v) linv[v] = 1.0f / lane_bcast(l, v + rowoff);

#pragma unroll
    for (int v = 0; v < 8; ++v) {
        const int rowg = i0 + v + rowoff;
        float* op = out + ((size_t)(bt * NN + rowg)) * FOUT;
        float h0 = acc0[v] * linv[v]; h0 = (h0 > 0.0f) ? h0 : (__expf(h0) - 1.0f);
        float h1 = acc1[v] * linv[v]; h1 = (h1 > 0.0f) ? h1 : (__expf(h1) - 1.0f);
        float h2 = acc2[v] * linv[v]; h2 = (h2 > 0.0f) ? h2 : (__expf(h2) - 1.0f);
        float h3 = acc3[v] * linv[v]; h3 = (h3 > 0.0f) ? h3 : (__expf(h3) - 1.0f);
        op[li]      = h0;
        op[16 + li] = h1;
        op[32 + li] = h2;
        op[48 + li] = h3;
    }
}

// =====================================================================
extern "C" void kernel_launch(void* const* d_in, const int* in_sizes, int n_in,
                              void* d_out, int out_size, void* d_ws, size_t ws_size,
                              hipStream_t stream) {
    const float* x    = (const float*)d_in[0];   // [4,12,1024,64]
    const int*   adj  = (const int*)  d_in[1];   // [1024,1024]
    const float* emb1 = (const float*)d_in[2];   // [1024,16]
    const float* emb2 = (const float*)d_in[3];   // [1024,16]
    const float* W    = (const float*)d_in[4];   // [64,64]
    const float* a    = (const float*)d_in[5];   // [128]
    const float* a2   = (const float*)d_in[6];   // [32]
    float* out = (float*)d_out;

    char* ws = (char*)d_ws;
    _Float16* whT     = (_Float16*)(ws + WS_WHT_OFF);
    float*    rterm   = (float*)   (ws + WS_RTERM_OFF);
    float*    cterm   = (float*)   (ws + WS_CTERM_OFF);
    unsigned* adjbits = (unsigned*)(ws + WS_ADJB_OFF);

    // K0: pack adjacency to bits (32x less mask traffic in K3).
    gat_adjpack_kernel<<<dim3((NN * 32) / 256), dim3(256), 0, stream>>>(adj, adjbits);

    // K1: projection -> WhT (f16). 49152 rows / 128 rows per block.
    gat_proj_kernel<<<dim3((BT * NN) / 128), dim3(256), 0, stream>>>(x, W, whT);

    // K2: score terms.
    gat_scores_kernel<<<dim3((BT * NN) / 256), dim3(256), 0, stream>>>(
        whT, emb1, emb2, a, a2, rterm, cterm);

    // K3: masked softmax attention + aggregation + ELU.
    // Dynamic LDS: 64 rows * LDS_STRIDE halves * 2 B = 132,096 B (WGP has 320 KB).
    gat_attn_kernel<<<dim3(NN / 128, BT), dim3(256),
                      (size_t)(FOUT * LDS_STRIDE * sizeof(_Float16)), stream>>>(
        adjbits, whT, rterm, cterm, out);
}
